// MambaFormerBlock_20289425506409
// MI455X (gfx1250) — compile-verified, matrix operand order checked by reading, not run on hardware
//
#include <hip/hip_runtime.h>

// MambaFormer block for gfx1250 (MI455X).
// B=2, L=1024, D=1024, H=16, HD=64, DIN=2048, DS=16, DTR=64, DCONV=4.

#define B_    2
#define L_    1024
#define D_    1024
#define H_    16
#define HD_   64
#define DIN_  2048
#define DS_   16
#define DTR_  64
#define QKVW_ 1152          // D + 2*HD
#define BL_   (B_ * L_)
#define EPS_  1e-5f

typedef __attribute__((ext_vector_type(16))) _Float16 v16h;
typedef __attribute__((ext_vector_type(8)))  float    v8f;

// ---------------------------------------------------------------------------
// Fragment loaders. CDNA5 16-bit A/B matrix layout (ISA 7.12.2):
//   lane l: row/col = l&15, K-halves: lanes 0-15 hold K=[0..7],[16..23],
//   lanes 16-31 hold K=[8..15],[24..31].  koff = (l>>4)*8, p = row + k0 + koff.
// ---------------------------------------------------------------------------
__device__ __forceinline__ v16h ldfrag(const float* __restrict__ p) {
  v16h r;
#pragma unroll
  for (int i = 0; i < 8; ++i) {
    r[i]     = (_Float16)p[i];
    r[i + 8] = (_Float16)p[i + 16];
  }
  return r;
}

__device__ __forceinline__ v16h ldfrag_h(const _Float16* p) {
  v16h r;
#pragma unroll
  for (int i = 0; i < 8; ++i) {
    r[i]     = p[i];
    r[i + 8] = p[i + 16];
  }
  return r;
}

__device__ __forceinline__ v8f wmma_f16(v16h a, v16h b, v8f c) {
  return __builtin_amdgcn_wmma_f32_16x16x32_f16(false, a, false, b, (short)0,
                                                c, false, false);
}

__device__ __forceinline__ float rmax16(float v) {
#pragma unroll
  for (int m = 1; m < 16; m <<= 1) v = fmaxf(v, __shfl_xor(v, m, 32));
  return v;
}
__device__ __forceinline__ float rsum16(float v) {
#pragma unroll
  for (int m = 1; m < 16; m <<= 1) v += __shfl_xor(v, m, 32);
  return v;
}

__device__ __forceinline__ float silu_f(float x) {
  return x / (1.f + __expf(-x));
}
__device__ __forceinline__ float softplus_f(float x) {
  return (x > 20.f) ? x : log1pf(__expf(x));
}

// ---------------------------------------------------------------------------
// Generic WMMA GEMM:  C[M,N](f32,ldc) = A[M,K](f32,lda) @ W[N,K]^T
//                     (+ bias[N]) (+ resid[M,N] stride N) (+ softplus)
// Wave tile: 32 (M) x 64 (N): 2 A-frags x 4 B-frags -> 8 WMMAs per k-step
// (A reused 4x, B 2x; ~1.5 KB of L2 traffic per WMMA). Tile id and nv are
// forced into SGPRs via readfirstlane so all guards are scalar branches.
// ---------------------------------------------------------------------------
__global__ __launch_bounds__(256) void gemm_wmma_kernel(
    const float* __restrict__ A, int lda,
    const float* __restrict__ W,
    const float* __restrict__ bias,
    const float* __restrict__ resid,
    float* __restrict__ C, int ldc,
    int M, int N, int K, int act) {
  const int wave = threadIdx.x >> 5;
  const int lane = threadIdx.x & 31;
  const int ntn  = (N + 63) >> 6;
  const int mt   = M >> 5;                              // 32-row tiles
  const int tile = __builtin_amdgcn_readfirstlane(
      blockIdx.x * (blockDim.x >> 5) + wave);
  if (tile >= mt * ntn) return;                         // scalar exit
  const int tm = tile / ntn;
  const int tn = tile % ntn;

  const int m    = lane & 15;
  const int half = lane >> 4;
  const int koff = half * 8;

  const float* arow0 = A + (size_t)(tm * 32 + m) * lda;
  const float* arow1 = A + (size_t)(tm * 32 + 16 + m) * lda;
  const float* wrow  = W + (size_t)(tn * 64 + m) * K;   // B-frag col = lane&15

  int nv = __builtin_amdgcn_readfirstlane((N - tn * 64) >> 4);
  if (nv > 4) nv = 4;                                   // valid 16-col subtiles

  v8f acc0[4] = {}, acc1[4] = {};
  if (nv == 4) {                                        // uniform fast path
    for (int k0 = 0; k0 < K; k0 += 32) {
      const v16h a0 = ldfrag(arow0 + k0 + koff);
      const v16h a1 = ldfrag(arow1 + k0 + koff);
#pragma unroll
      for (int j = 0; j < 4; ++j) {
        const v16h bf = ldfrag(wrow + (size_t)(j * 16) * K + k0 + koff);
        acc0[j] = wmma_f16(a0, bf, acc0[j]);
        acc1[j] = wmma_f16(a1, bf, acc1[j]);
      }
    }
  } else {                                              // edge tiles (N=96)
    for (int k0 = 0; k0 < K; k0 += 32) {
      const v16h a0 = ldfrag(arow0 + k0 + koff);
      const v16h a1 = ldfrag(arow1 + k0 + koff);
#pragma unroll
      for (int j = 0; j < 4; ++j) {
        if (j >= nv) break;                             // scalar branch (nv in SGPR)
        const v16h bf = ldfrag(wrow + (size_t)(j * 16) * K + k0 + koff);
        acc0[j] = wmma_f16(a0, bf, acc0[j]);
        acc1[j] = wmma_f16(a1, bf, acc1[j]);
      }
    }
  }

#pragma unroll
  for (int j = 0; j < 4; ++j) {
    if (j >= nv) break;
    const int col = tn * 64 + j * 16 + m;
#pragma unroll
    for (int r = 0; r < 8; ++r) {
      const int row0 = tm * 32 + r + 8 * half;   // C layout: VGPR r -> M=r+8*(l>>4)
      const int row1 = row0 + 16;
      float v0 = acc0[j][r];
      float v1 = acc1[j][r];
      if (bias) { const float bv = bias[col]; v0 += bv; v1 += bv; }
      if (resid) {
        v0 += resid[(size_t)row0 * N + col];
        v1 += resid[(size_t)row1 * N + col];
      }
      if (act == 1) { v0 = softplus_f(v0); v1 = softplus_f(v1); }
      C[(size_t)row0 * ldc + col] = v0;
      C[(size_t)row1 * ldc + col] = v1;
    }
  }
}

// ---------------------------------------------------------------------------
// Dual LayerNorm over last dim (two affine pairs share one mean/var pass).
// One block per row.
// ---------------------------------------------------------------------------
__global__ __launch_bounds__(256) void ln2_kernel(
    const float* __restrict__ x,
    const float* __restrict__ w1, const float* __restrict__ b1,
    float* __restrict__ o1,
    const float* __restrict__ w2, const float* __restrict__ b2,
    float* __restrict__ o2, int D) {
  __shared__ float ssum[256], ssq[256];
  const int row = blockIdx.x;
  const float* xr = x + (size_t)row * D;
  float s = 0.f, q = 0.f;
  for (int i = threadIdx.x; i < D; i += 256) {
    float v = xr[i]; s += v; q += v * v;
  }
  ssum[threadIdx.x] = s; ssq[threadIdx.x] = q;
  __syncthreads();
  for (int st = 128; st > 0; st >>= 1) {
    if ((int)threadIdx.x < st) {
      ssum[threadIdx.x] += ssum[threadIdx.x + st];
      ssq[threadIdx.x]  += ssq[threadIdx.x + st];
    }
    __syncthreads();
  }
  const float mean = ssum[0] / (float)D;
  const float var  = ssq[0] / (float)D - mean * mean;
  const float rstd = rsqrtf(var + EPS_);
  for (int i = threadIdx.x; i < D; i += 256) {
    const float xc = (xr[i] - mean) * rstd;
    o1[(size_t)row * D + i] = xc * w1[i] + b1[i];
    if (o2) o2[(size_t)row * D + i] = xc * w2[i] + b2[i];
  }
}

// ---------------------------------------------------------------------------
// Depthwise causal conv (DCONV=4) + bias + SiLU over u = xr[..., :DIN].
// ---------------------------------------------------------------------------
__global__ __launch_bounds__(256) void conv_silu_kernel(
    const float* __restrict__ xr, const float* __restrict__ cw,
    const float* __restrict__ cb, float* __restrict__ u) {
  const int idx = blockIdx.x * blockDim.x + threadIdx.x;
  if (idx >= B_ * L_ * DIN_) return;
  const int c = idx % DIN_;
  const int l = (idx / DIN_) % L_;
  const int b = idx / (DIN_ * L_);
  const float* base = xr + (size_t)b * L_ * (2 * DIN_);
  float s = cb[c];
#pragma unroll
  for (int t = 0; t < 4; ++t) {
    const int ll = l - 3 + t;
    if (ll >= 0) s += cw[c * 4 + t] * base[(size_t)ll * (2 * DIN_) + c];
  }
  u[idx] = silu_f(s);
}

// ---------------------------------------------------------------------------
// Selective scan: thread per (b,d) channel, DS=16 states in registers,
// sequential over L. Epilogue folds D_skip and y *= silu(res).
// dbc row = [dt(64) | B(16) | C(16)], stride 96.
// ---------------------------------------------------------------------------
__global__ __launch_bounds__(256) void scan_kernel(
    const float* __restrict__ delta, const float* __restrict__ u,
    const float* __restrict__ dbc, const float* __restrict__ xr,
    const float* __restrict__ A_log, const float* __restrict__ D_skip,
    float* __restrict__ y) {
  const int idx = blockIdx.x * blockDim.x + threadIdx.x;
  if (idx >= B_ * DIN_) return;
  const int d = idx % DIN_;
  const int b = idx / DIN_;
  float Am[DS_], h[DS_];
#pragma unroll
  for (int n = 0; n < DS_; ++n) {
    Am[n] = -__expf(A_log[(size_t)d * DS_ + n]);
    h[n]  = 0.f;
  }
  const float dsk = D_skip[d];
  for (int l = 0; l < L_; ++l) {
    const size_t t  = (size_t)b * L_ + l;
    const float  dl = delta[t * DIN_ + d];
    const float  uv = u[t * DIN_ + d];
    const float* bc = dbc + t * 96;
    const float  du = dl * uv;
    float acc = 0.f;
#pragma unroll
    for (int n = 0; n < DS_; ++n) {
      h[n] = __expf(dl * Am[n]) * h[n] + du * bc[64 + n];
      acc += h[n] * bc[80 + n];
    }
    const float r = xr[t * (2 * DIN_) + DIN_ + d];   // res half of in_proj output
    y[t * DIN_ + d] = (acc + uv * dsk) * silu_f(r);
  }
}

// ---------------------------------------------------------------------------
// Causal MQA flash attention, all-WMMA. One wave per (b, h, 16-query tile).
// qkv rows: [q(1024) | k(64) | v(64)], stride QKVW_.
// S = Q·K^T via 2 chained 16x16x32 f16 WMMAs (HD=64); online softmax with
// 16-lane shuffles; P bounced through LDS into A-frag layout; O += P·V.
// ---------------------------------------------------------------------------
__global__ __launch_bounds__(32) void attn_kernel(
    const float* __restrict__ qkv, float* __restrict__ out) {
  __shared__ _Float16 Pl[16 * 32];

  const int lane = threadIdx.x & 31;
  const int tile = blockIdx.x;
  const int qt   = tile & (L_ / 16 - 1);      // 64 q-tiles
  const int h    = (tile >> 6) & (H_ - 1);
  const int b    = tile >> 10;
  const int q0   = qt * 16;

  const int m    = lane & 15;
  const int half = lane >> 4;
  const int koff = half * 8;

  const float* qrow = qkv + ((size_t)(b * L_) + q0 + m) * QKVW_ + h * HD_;
  const v16h aQ0 = ldfrag(qrow + 0  + koff);
  const v16h aQ1 = ldfrag(qrow + 32 + koff);

  v8f oA0 = {}, oA1 = {}, oA2 = {}, oA3 = {};
  float rm[8], rs[8];
#pragma unroll
  for (int r = 0; r < 8; ++r) { rm[r] = -1e30f; rs[r] = 0.f; }

  const int kend = q0 + 16;                   // keys needed: 0 .. q0+15
  for (int kb = 0; kb < kend; kb += 32) {
    // ---- S = (Q K^T) / 8 for two 16-key subtiles -------------------------
    const float* krow0 = qkv + ((size_t)(b * L_) + kb + m)      * QKVW_ + D_;
    const float* krow1 = qkv + ((size_t)(b * L_) + kb + 16 + m) * QKVW_ + D_;
    v8f s0 = {}, s1 = {};
    {
      v16h bk = ldfrag(krow0 + 0 + koff);
      s0 = wmma_f16(aQ0, bk, s0);
      bk = ldfrag(krow0 + 32 + koff);
      s0 = wmma_f16(aQ1, bk, s0);
      bk = ldfrag(krow1 + 0 + koff);
      s1 = wmma_f16(aQ0, bk, s1);
      bk = ldfrag(krow1 + 32 + koff);
      s1 = wmma_f16(aQ1, bk, s1);
    }
    const bool need_mask = (kb + 31 > q0);
#pragma unroll
    for (int r = 0; r < 8; ++r) {
      float v0 = s0[r] * 0.125f;              // 1/sqrt(HD)
      float v1 = s1[r] * 0.125f;
      if (need_mask) {
        const int qg = q0 + r + 8 * half;
        if (kb + m      > qg) v0 = -1e30f;
        if (kb + 16 + m > qg) v1 = -1e30f;
      }
      s0[r] = v0; s1[r] = v1;
    }

    // ---- online softmax update ------------------------------------------
    __syncthreads();                          // LDS reuse ordering across iters
#pragma unroll
    for (int r = 0; r < 8; ++r) {
      float bm    = rmax16(fmaxf(s0[r], s1[r]));
      float mnew  = fmaxf(rm[r], bm);
      float alpha = __expf(rm[r] - mnew);
      rm[r] = mnew;
      float p0 = __expf(s0[r] - mnew);
      float p1 = __expf(s1[r] - mnew);
      rs[r] = rs[r] * alpha + rsum16(p0 + p1);
      Pl[(r + 8 * half) * 32 + m]      = (_Float16)p0;
      Pl[(r + 8 * half) * 32 + 16 + m] = (_Float16)p1;
      oA0[r] *= alpha; oA1[r] *= alpha; oA2[r] *= alpha; oA3[r] *= alpha;
    }
    __syncthreads();

    // ---- O += P (16x32) @ V (32xHD) -------------------------------------
    const v16h aP = ldfrag_h(&Pl[m * 32 + koff]);
    const float* vbase = qkv + ((size_t)(b * L_) + kb) * QKVW_ + D_ + HD_;
#pragma unroll
    for (int dt = 0; dt < 4; ++dt) {
      v16h bV;
#pragma unroll
      for (int i = 0; i < 8; ++i) {
        bV[i]     = (_Float16)vbase[(size_t)(koff + i)      * QKVW_ + dt * 16 + m];
        bV[i + 8] = (_Float16)vbase[(size_t)(16 + koff + i) * QKVW_ + dt * 16 + m];
      }
      v8f* po = (dt == 0) ? &oA0 : (dt == 1) ? &oA1 : (dt == 2) ? &oA2 : &oA3;
      *po = wmma_f16(aP, bV, *po);
    }
  }

  // ---- normalize and write (B,L,H,HD) -> (B,L,D) -------------------------
#pragma unroll
  for (int dt = 0; dt < 4; ++dt) {
    const v8f* po = (dt == 0) ? &oA0 : (dt == 1) ? &oA1 : (dt == 2) ? &oA2 : &oA3;
#pragma unroll
    for (int r = 0; r < 8; ++r) {
      const int row = q0 + r + 8 * half;
      out[((size_t)(b * L_) + row) * D_ + h * HD_ + dt * 16 + m] = (*po)[r] / rs[r];
    }
  }
}

// ---------------------------------------------------------------------------
// Launcher
// ---------------------------------------------------------------------------
static inline int gemm_blocks(int M, int N) {
  const int tiles = (M >> 5) * ((N + 63) >> 6);
  return (tiles + 7) / 8;   // 8 waves / 256-thread block
}

extern "C" void kernel_launch(void* const* d_in, const int* in_sizes, int n_in,
                              void* d_out, int out_size, void* d_ws, size_t ws_size,
                              hipStream_t stream) {
  (void)in_sizes; (void)n_in; (void)out_size; (void)ws_size;

  const float* x          = (const float*)d_in[0];
  const float* mnorm_w    = (const float*)d_in[1];
  const float* mnorm_b    = (const float*)d_in[2];
  const float* in_proj_w  = (const float*)d_in[3];
  const float* conv_w     = (const float*)d_in[4];
  const float* conv_b     = (const float*)d_in[5];
  const float* x_proj_w   = (const float*)d_in[6];
  const float* dt_proj_w  = (const float*)d_in[7];
  const float* dt_proj_b  = (const float*)d_in[8];
  const float* A_log      = (const float*)d_in[9];
  const float* D_skip     = (const float*)d_in[10];
  const float* out_proj_w = (const float*)d_in[11];
  const float* norm1_w    = (const float*)d_in[12];
  const float* norm1_b    = (const float*)d_in[13];
  const float* wqkv_w     = (const float*)d_in[14];
  const float* wqkv_b     = (const float*)d_in[15];
  const float* oattn_w    = (const float*)d_in[16];
  const float* oattn_b    = (const float*)d_in[17];
  const float* fuse_w     = (const float*)d_in[18];
  const float* fuse_b     = (const float*)d_in[19];
  const float* fln_w      = (const float*)d_in[20];
  const float* fln_b      = (const float*)d_in[21];
  float* outp             = (float*)d_out;

  char* wsp = (char*)d_ws;
  auto carve = [&](size_t nfloats) -> float* {
    float* p = (float*)wsp;
    wsp += nfloats * sizeof(float);
    return p;
  };
  float* xnM       = carve((size_t)BL_ * D_);        // LN(mnorm)
  float* xnA       = carve((size_t)BL_ * D_);        // LN(norm1)
  float* xr        = carve((size_t)BL_ * 2 * DIN_);  // in_proj out [u | res]
  float* u         = carve((size_t)BL_ * DIN_);      // conv+silu
  float* dbc       = carve((size_t)BL_ * 96);        // x_proj out
  float* delta     = carve((size_t)BL_ * DIN_);      // softplus(dt_proj)
  float* y         = carve((size_t)BL_ * DIN_);      // scan out (*silu(res))
  float* qkv       = carve((size_t)BL_ * QKVW_);
  float* o_attn    = carve((size_t)BL_ * D_);
  float* combined  = carve((size_t)BL_ * 2 * D_);    // [mamba_out | attn_out]
  float* fused_pre = carve((size_t)BL_ * D_);

  // 1) both LayerNorms of x in one pass
  ln2_kernel<<<BL_, 256, 0, stream>>>(x, mnorm_w, mnorm_b, xnM,
                                      norm1_w, norm1_b, xnA, D_);

  // 2) xr = xnM @ in_proj_w^T                      (2048 x 4096 x 1024)
  gemm_wmma_kernel<<<gemm_blocks(BL_, 2 * DIN_), 256, 0, stream>>>(
      xnM, D_, in_proj_w, nullptr, nullptr, xr, 2 * DIN_, BL_, 2 * DIN_, D_, 0);

  // 3) u = silu(depthwise causal conv(xr[..., :DIN]) + conv_b)
  conv_silu_kernel<<<(B_ * L_ * DIN_ + 255) / 256, 256, 0, stream>>>(
      xr, conv_w, conv_b, u);

  // 4) dbc = u @ x_proj_w^T                        (2048 x 96 x 2048)
  gemm_wmma_kernel<<<gemm_blocks(BL_, 96), 256, 0, stream>>>(
      u, DIN_, x_proj_w, nullptr, nullptr, dbc, 96, BL_, 96, DIN_, 0);

  // 5) delta = softplus(dbc[:, :64] @ dt_proj_w^T + dt_proj_b)
  gemm_wmma_kernel<<<gemm_blocks(BL_, DIN_), 256, 0, stream>>>(
      dbc, 96, dt_proj_w, dt_proj_b, nullptr, delta, DIN_, BL_, DIN_, DTR_, 1);

  // 6) selective scan (+ D_skip skip, * silu(res))
  scan_kernel<<<(B_ * DIN_ + 255) / 256, 256, 0, stream>>>(
      delta, u, dbc, xr, A_log, D_skip, y);

  // 7) mamba_out = y @ out_proj_w^T + x  -> combined[:, :D]
  gemm_wmma_kernel<<<gemm_blocks(BL_, D_), 256, 0, stream>>>(
      y, DIN_, out_proj_w, nullptr, x, combined, 2 * D_, BL_, D_, DIN_, 0);

  // 8) qkv = xnA @ wqkv_w^T + wqkv_b              (2048 x 1152 x 1024)
  gemm_wmma_kernel<<<gemm_blocks(BL_, QKVW_), 256, 0, stream>>>(
      xnA, D_, wqkv_w, wqkv_b, nullptr, qkv, QKVW_, BL_, QKVW_, D_, 0);

  // 9) causal MQA flash attention -> o_attn
  attn_kernel<<<B_ * H_ * (L_ / 16), 32, 0, stream>>>(qkv, o_attn);

  // 10) attn_out = x + o_attn @ oattn_w^T + oattn_b  -> combined[:, D:]
  gemm_wmma_kernel<<<gemm_blocks(BL_, D_), 256, 0, stream>>>(
      o_attn, D_, oattn_w, oattn_b, x, combined + D_, 2 * D_, BL_, D_, D_, 0);

  // 11) fused_pre = combined @ fuse_w^T + fuse_b   (2048 x 1024 x 2048)
  gemm_wmma_kernel<<<gemm_blocks(BL_, D_), 256, 0, stream>>>(
      combined, 2 * D_, fuse_w, fuse_b, nullptr, fused_pre, D_, BL_, D_, 2 * D_, 0);

  // 12) out = LN(fused_pre; fln)
  ln2_kernel<<<BL_, 256, 0, stream>>>(fused_pre, fln_w, fln_b, outp,
                                      nullptr, nullptr, nullptr, D_);
}
